// EFH_28389733827091
// MI455X (gfx1250) — matrix operand order repaired and checked by reading, not compiled, not run on hardware
//
#include <hip/hip_runtime.h>
#include <stdint.h>

typedef __attribute__((ext_vector_type(16))) _Float16 v16h;
typedef __attribute__((ext_vector_type(8)))  float    v8f;

#define NB    32      // batch
#define C_IN  2       // visible channels
#define C_OUT 20      // hidden channels
#define HW    65536   // 256*256
#define KPAD  32      // channel dim padded to WMMA K
#define HS_ST 33      // LDS row stride for channel vectors (bank-conflict free)
#define EH_ST 21      // LDS row stride for eta_hid

// ---- stateless counter-based PRNG (deterministic across replays) ----
__device__ __forceinline__ unsigned mix32(unsigned x) {
  x ^= x >> 16; x *= 0x7feb352dU;
  x ^= x >> 15; x *= 0x846ca68bU;
  x ^= x >> 16;
  return x;
}
__device__ __forceinline__ float uniform01(unsigned pix, unsigned salt) {
  unsigned h = mix32(pix * 0x9E3779B9u ^ (salt * 0x85EBCA6Bu + 0x6546B64u));
  return (float)(h >> 8) * (1.0f / 16777216.0f);   // [0,1)
}

__global__ __launch_bounds__(256) void rbm_cd_fused(
    const float* __restrict__ hid_in,   // [32,20,256,256]
    const float* __restrict__ Wf,       // [20,2,1,1]
    const float* __restrict__ b_vis,    // [2]
    const float* __restrict__ b_hid,    // [20]
    const int*   __restrict__ n_steps_p,
    float* __restrict__ out)            // mu_vis | vis | mu_hid | hid (flat)
{
  __shared__ float Hs[256 * HS_ST];     // per-pixel channel vector (hid, then vis), zero-padded to 32
  __shared__ float Ev[256 * C_IN];      // eta_vis
  __shared__ float Eh[256 * EH_ST];     // eta_hid
  __shared__ float Wl[C_OUT * C_IN];
  __shared__ float Bv[C_IN];
  __shared__ float Bh[C_OUT];

  const int tid      = threadIdx.x;
  const int lane     = tid & 31;
  const int waveBase = tid & ~31;       // first pixel row (within block) of this wave
  const int h16      = lane >> 4;       // half-wave select
  const int n16      = lane & 15;

  if (tid < C_OUT * C_IN) Wl[tid] = Wf[tid];
  if (tid < C_IN)         Bv[tid] = b_vis[tid];
  if (tid < C_OUT)        Bh[tid] = b_hid[tid];
  __syncthreads();

  // ---- constant B operands (32x16) in this code's contiguous k order (A uses the same):
  // lane L, VGPR v holds K = 16*(L>>4) + 2v (+1), column N = L&15.
  v16h Bemit, Binf0, Binf1;
#pragma unroll
  for (int v = 0; v < 8; ++v) {
    const int K0 = 16 * h16 + 2 * v;
    const int K1 = K0 + 1;
    // emit:  K = o (0..19), N = c (0..1):  B[o][c] = W[o,c]
    Bemit[2*v]   = (_Float16)((K0 < C_OUT && n16 < C_IN) ? Wl[K0 * C_IN + n16] : 0.0f);
    Bemit[2*v+1] = (_Float16)((K1 < C_OUT && n16 < C_IN) ? Wl[K1 * C_IN + n16] : 0.0f);
    // infer block 0: K = c (0..1), N = o (0..15): B[c][o] = W[o,c]
    Binf0[2*v]   = (_Float16)((K0 < C_IN) ? Wl[n16 * C_IN + K0] : 0.0f);
    Binf0[2*v+1] = (_Float16)((K1 < C_IN) ? Wl[n16 * C_IN + K1] : 0.0f);
    // infer block 1: N = o = 16..19
    Binf1[2*v]   = (_Float16)((K0 < C_IN && n16 < (C_OUT - 16)) ? Wl[(16 + n16) * C_IN + K0] : 0.0f);
    Binf1[2*v+1] = (_Float16)((K1 < C_IN && n16 < (C_OUT - 16)) ? Wl[(16 + n16) * C_IN + K1] : 0.0f);
  }

  const unsigned pix = blockIdx.x * 256u + (unsigned)tid;  // global pixel id
  const int nImg = (int)(pix >> 16);                       // batch index (HW = 2^16)
  const int rem  = (int)(pix & (HW - 1));                  // h*W + w

  // ---- load this pixel's 20 hid channels into LDS (coalesced per channel plane)
  float* myRow = &Hs[tid * HS_ST];
  {
    const float* base = hid_in + (size_t)nImg * C_OUT * HW + rem;
#pragma unroll
    for (int o = 0; o < C_OUT; ++o) myRow[o] = base[(size_t)o * HW];
#pragma unroll
    for (int o = C_OUT; o < KPAD; ++o) myRow[o] = 0.0f;
  }
  __syncthreads();

  const int n_steps = *n_steps_p;       // uniform scalar load

  const size_t oMuVis = 0;
  const size_t oVis   = (size_t)NB * C_IN * HW;
  const size_t oMuHid = 2 * oVis;
  const size_t oHid   = oMuHid + (size_t)NB * C_OUT * HW;

  for (int step = 0; step < n_steps; ++step) {
    // ================= emit: eta_vis = hid x W  (K=20→32, N=2 of 16) =================
#pragma unroll
    for (int grp = 0; grp < 2; ++grp) {
      // A (16x32) in this code's contiguous k order, same as B: lane = (M&15) + 16*(K>>4); VGPR v <-> K pair 16*(L>>4) + 2v
      const float* Arow = &Hs[(waveBase + grp * 16 + n16) * HS_ST];
      v16h A;
#pragma unroll
      for (int v = 0; v < 8; ++v) {
        const int K0 = 16 * h16 + 2 * v;
        A[2*v]   = (_Float16)Arow[K0];
        A[2*v+1] = (_Float16)Arow[K0 + 1];
      }
      v8f acc = {};
      acc = __builtin_amdgcn_wmma_f32_16x16x32_f16(false, A, false, Bemit,
                                                   (short)0, acc, false, false);
      // D layout: lane holds (M = 8*h16 + r, N = n16); only N < C_IN matters
      if (n16 < C_IN) {
#pragma unroll
        for (int r = 0; r < 8; ++r)
          Ev[(waveBase + grp * 16 + 8 * h16 + r) * C_IN + n16] = acc[r];
      }
    }
    __syncthreads();

    // ================= visible: sigmoid + Bernoulli =================
    const float e0  = Ev[tid * C_IN + 0] + Bv[0];
    const float e1  = Ev[tid * C_IN + 1] + Bv[1];
    const float mu0 = 1.0f / (1.0f + __expf(-e0));
    const float mu1 = 1.0f / (1.0f + __expf(-e1));
    const float s0  = (uniform01(pix, ((unsigned)step << 8) + 0u) < mu0) ? 1.0f : 0.0f;
    const float s1  = (uniform01(pix, ((unsigned)step << 8) + 1u) < mu1) ? 1.0f : 0.0f;

    myRow[0] = s0; myRow[1] = s1;
#pragma unroll
    for (int o = C_IN; o < KPAD; ++o) myRow[o] = 0.0f;
    __syncthreads();

    // ================= infer: eta_hid = vis x W^T (K=2→32, N=20 split 16+4) =================
#pragma unroll
    for (int grp = 0; grp < 2; ++grp) {
      const float* Arow = &Hs[(waveBase + grp * 16 + n16) * HS_ST];
      v16h A;
#pragma unroll
      for (int v = 0; v < 8; ++v) {
        const int K0 = 16 * h16 + 2 * v;
        A[2*v]   = (_Float16)Arow[K0];
        A[2*v+1] = (_Float16)Arow[K0 + 1];
      }
      v8f d0 = {}, d1 = {};
      d0 = __builtin_amdgcn_wmma_f32_16x16x32_f16(false, A, false, Binf0,
                                                  (short)0, d0, false, false);
      d1 = __builtin_amdgcn_wmma_f32_16x16x32_f16(false, A, false, Binf1,
                                                  (short)0, d1, false, false);
#pragma unroll
      for (int r = 0; r < 8; ++r) {
        const int prow = waveBase + grp * 16 + 8 * h16 + r;
        Eh[prow * EH_ST + n16] = d0[r];
        if (n16 < (C_OUT - 16)) Eh[prow * EH_ST + 16 + n16] = d1[r];
      }
    }
    __syncthreads();

    // ================= hidden: softmax(20), mu_hid = 4*softmax, multinomial(4) =================
    float eta[C_OUT];
    float mx = -3.4e38f;
#pragma unroll
    for (int o = 0; o < C_OUT; ++o) {
      eta[o] = Eh[tid * EH_ST + o] + Bh[o];
      mx = fmaxf(mx, eta[o]);
    }
    float sum = 0.0f;
#pragma unroll
    for (int o = 0; o < C_OUT; ++o) { eta[o] = __expf(eta[o] - mx); sum += eta[o]; }
    const float inv = 1.0f / sum;

    int idx0, idx1, idx2, idx3;
    {
      int sel[4] = {0, 0, 0, 0};
      float u[4];
#pragma unroll
      for (int t = 0; t < 4; ++t) u[t] = uniform01(pix, ((unsigned)step << 8) + 16u + (unsigned)t);
      float cdf = 0.0f;
#pragma unroll
      for (int o = 0; o < C_OUT; ++o) {
        cdf += eta[o] * inv;              // cumsum -> searchsorted
#pragma unroll
        for (int t = 0; t < 4; ++t) sel[t] += (u[t] >= cdf) ? 1 : 0;
      }
      idx0 = sel[0] < C_OUT - 1 ? sel[0] : C_OUT - 1;
      idx1 = sel[1] < C_OUT - 1 ? sel[1] : C_OUT - 1;
      idx2 = sel[2] < C_OUT - 1 ? sel[2] : C_OUT - 1;
      idx3 = sel[3] < C_OUT - 1 ? sel[3] : C_OUT - 1;
    }

    const bool last = (step == n_steps - 1);
    if (last) {
      out[oMuVis + ((size_t)nImg * C_IN + 0) * HW + rem] = mu0;
      out[oMuVis + ((size_t)nImg * C_IN + 1) * HW + rem] = mu1;
      out[oVis   + ((size_t)nImg * C_IN + 0) * HW + rem] = s0;
      out[oVis   + ((size_t)nImg * C_IN + 1) * HW + rem] = s1;
    }
#pragma unroll
    for (int o = 0; o < C_OUT; ++o) {
      const float cnt = (float)((idx0 == o) + (idx1 == o) + (idx2 == o) + (idx3 == o));
      myRow[o] = cnt;                     // new hid state for next CD step
      if (last) {
        out[oMuHid + ((size_t)nImg * C_OUT + o) * HW + rem] = 4.0f * eta[o] * inv;  // N_TRIALS * softmax
        out[oHid   + ((size_t)nImg * C_OUT + o) * HW + rem] = cnt;
      }
    }
    __syncthreads();
  }
}

extern "C" void kernel_launch(void* const* d_in, const int* in_sizes, int n_in,
                              void* d_out, int out_size, void* d_ws, size_t ws_size,
                              hipStream_t stream) {
  (void)in_sizes; (void)n_in; (void)out_size; (void)d_ws; (void)ws_size;
  const float* hid  = (const float*)d_in[0];
  const float* W    = (const float*)d_in[1];
  const float* bvis = (const float*)d_in[2];
  const float* bhid = (const float*)d_in[3];
  const int*   nstp = (const int*)d_in[4];
  float* out = (float*)d_out;

  const int R = NB * HW;                 // 2,097,152 pixels, one per thread
  rbm_cd_fused<<<R / 256, 256, 0, stream>>>(hid, W, bvis, bhid, nstp, out);
}
